// HierarchicalCircuitV4_27144193310741
// MI455X (gfx1250) — matrix operand order repaired and checked by reading, not compiled, not run on hardware
//
#include <hip/hip_runtime.h>
#include <hip/hip_bf16.h>
#include <math.h>

// ---------------- problem constants ----------------
#define KSLOT   16
#define NSTEPS  24
#define NSLOT   256          // G*K
#define MAXNODE 50
#define BATCH   32
#define SEQL    2048
#define T_REAL  (BATCH * (SEQL + 1))   // 65568 tokens (edge + query per batch)
#define TPAD    65664                  // multiple of 128 (GEMM block-M)
#define SROWS   (BATCH * NSLOT)        // 8192 slot rows (multiple of 128)

typedef __attribute__((ext_vector_type(16))) __bf16 v16bf;
typedef __attribute__((ext_vector_type(8)))  float  v8f;
typedef __attribute__((ext_vector_type(4)))  int    v4i;

__device__ __forceinline__ float bf2f(unsigned short h) {
    return __uint_as_float(((unsigned int)h) << 16);
}
__device__ __forceinline__ unsigned short f2bf(float f) {
    unsigned int u = __float_as_uint(f);
    u += 0x7FFFu + ((u >> 16) & 1u);      // round-to-nearest-even
    return (unsigned short)(u >> 16);
}
__device__ __forceinline__ float gelu_exact(float x) {
    return 0.5f * x * (1.0f + erff(x * 0.70710678118654752440f));
}

// ---- gfx1250 async global->LDS staging (guarded; sync fallback) ----
#if defined(__has_builtin)
#  if __has_builtin(__builtin_amdgcn_global_load_async_to_lds_b128)
#    define USE_ASYNC_LDS 1
#  endif
#  if __has_builtin(__builtin_amdgcn_s_wait_asynccnt)
#    define HAVE_WAIT_ASYNC 1
#  endif
#endif

#if defined(USE_ASYNC_LDS)
__device__ __forceinline__ void async_b128(const unsigned short* g, unsigned short* l) {
    // prototype: (v4i addrspace(1)*, v4i addrspace(3)*, imm offset, imm cpol)
    __builtin_amdgcn_global_load_async_to_lds_b128(
        (__attribute__((address_space(1))) v4i*)(g),
        (__attribute__((address_space(3))) v4i*)(l), 0, 0);
}
__device__ __forceinline__ void wait_async_le3() {
#if defined(HAVE_WAIT_ASYNC)
    __builtin_amdgcn_s_wait_asynccnt(3);
#else
    asm volatile("s_wait_asynccnt 0x3" ::: "memory");
#endif
}
__device__ __forceinline__ void wait_async_le0() {
#if defined(HAVE_WAIT_ASYNC)
    __builtin_amdgcn_s_wait_asynccnt(0);
#else
    asm volatile("s_wait_asynccnt 0x0" ::: "memory");
#endif
}
#endif

// ======================================================================
// Tiled bf16 WMMA GEMM:  C[M,N] = epilogue(A[M,K] * W[N,K]^T + bias[N])
//   block tile 128x64, 8 wave32 waves, each wave -> 32(m) x 32(n)
//   -> 4 x v_wmma_f32_16x16x32_bf16 per wave per K-step of 32
//   CONTRACT: M % 128 == 0, N % 64 == 0, K % 32 == 0 (buffers padded)
// ======================================================================
#define SH_LD 40   // padded halves per LDS row (conflict-free b128 reads)

template <bool ACT_GELU, bool OUT_BF16>
__global__ __launch_bounds__(256) void gemm_bf16_wmma(
    const unsigned short* __restrict__ A, int lda,
    const unsigned short* __restrict__ W, int ldb,
    const float* __restrict__ bias,
    void* __restrict__ Cout, int ldc,
    int M, int N, int K)
{
    // double buffers (buffer 1 unused on the sync fallback path)
    __shared__ __align__(16) unsigned short shA[2 * 128 * SH_LD];
    __shared__ __align__(16) unsigned short shB[2 * 64 * SH_LD];

    const int tid  = threadIdx.x;
    const int lane = tid & 31;
    const int wave = tid >> 5;
    const int wm   = wave & 3;    // 0..3 : 32-row strip
    const int wn   = wave >> 2;   // 0..1 : 32-col strip
    const int mBlk = blockIdx.x * 128;
    const int nBlk = blockIdx.y * 64;

    const int aRow = tid >> 2;    // 0..63
    const int aSeg = tid & 3;     // 4 x 8 halves per 32-half row

    const int r  = lane & 15;     // row (A) / col (B) within 16-tile
    const int hi = lane >> 4;

    v8f acc00 = {}, acc01 = {}, acc10 = {}, acc11 = {};

#if defined(USE_ASYNC_LDS)
    // ---- async double-buffered pipeline ----
    {
        // stage kk=0 into buffer 0 (3 async b128 per wave: A lo, A hi, B)
        async_b128(A + (size_t)(mBlk + aRow) * lda + aSeg * 8,
                   shA + aRow * SH_LD + aSeg * 8);
        async_b128(A + (size_t)(mBlk + 64 + aRow) * lda + aSeg * 8,
                   shA + (64 + aRow) * SH_LD + aSeg * 8);
        async_b128(W + (size_t)(nBlk + aRow) * ldb + aSeg * 8,
                   shB + aRow * SH_LD + aSeg * 8);
    }
    int cur = 0;
    for (int kk = 0; kk < K; kk += 32) {
        const bool hasNext = (kk + 32 < K);
        if (hasNext) {
            const int nb = cur ^ 1;
            unsigned short* bA = shA + nb * (128 * SH_LD);
            unsigned short* bB = shB + nb * (64 * SH_LD);
            async_b128(A + (size_t)(mBlk + aRow) * lda + (kk + 32) + aSeg * 8,
                       bA + aRow * SH_LD + aSeg * 8);
            async_b128(A + (size_t)(mBlk + 64 + aRow) * lda + (kk + 32) + aSeg * 8,
                       bA + (64 + aRow) * SH_LD + aSeg * 8);
            async_b128(W + (size_t)(nBlk + aRow) * ldb + (kk + 32) + aSeg * 8,
                       bB + aRow * SH_LD + aSeg * 8);
            wait_async_le3();   // current buffer's group complete (in-order)
        } else {
            wait_async_le0();
        }
        __syncthreads();        // all waves' async writes for `cur` visible

        const unsigned short* bA = shA + cur * (128 * SH_LD);
        const unsigned short* bB = shB + cur * (64 * SH_LD);
#else
    for (int kk = 0; kk < K; kk += 32) {
        // ---- sync staging: global -> regs -> LDS (buffer 0) ----
        uint4 va0 = *(const uint4*)(A + (size_t)(mBlk + aRow) * lda + kk + aSeg * 8);
        uint4 va1 = *(const uint4*)(A + (size_t)(mBlk + 64 + aRow) * lda + kk + aSeg * 8);
        uint4 vb  = *(const uint4*)(W + (size_t)(nBlk + aRow) * ldb + kk + aSeg * 8);
        *(uint4*)(&shA[aRow * SH_LD + aSeg * 8])        = va0;
        *(uint4*)(&shA[(64 + aRow) * SH_LD + aSeg * 8]) = va1;
        *(uint4*)(&shB[aRow * SH_LD + aSeg * 8])        = vb;
        __syncthreads();
        const unsigned short* bA = shA;
        const unsigned short* bB = shB;
#endif
        // ---- fragments (ISA 16-bit A 16x32 / B 32x16 layouts) ----
        union { v16bf v; uint4 u[2]; } fa0, fa1, fb0, fb1;
        // A: lane<16 -> K{0..7,16..23}, lane>=16 -> K{8..15,24..31}
        fa0.u[0] = *(const uint4*)(&bA[(wm * 32 + r) * SH_LD + hi * 8]);
        fa0.u[1] = *(const uint4*)(&bA[(wm * 32 + r) * SH_LD + 16 + hi * 8]);
        fa1.u[0] = *(const uint4*)(&bA[(wm * 32 + 16 + r) * SH_LD + hi * 8]);
        fa1.u[1] = *(const uint4*)(&bA[(wm * 32 + 16 + r) * SH_LD + 16 + hi * 8]);
        // B: lane<16 -> K{0..7,16..23}, lane>=16 -> K{8..15,24..31}
        fb0.u[0] = *(const uint4*)(&bB[(wn * 32 + r) * SH_LD + hi * 8]);
        fb0.u[1] = *(const uint4*)(&bB[(wn * 32 + r) * SH_LD + 16 + hi * 8]);
        fb1.u[0] = *(const uint4*)(&bB[(wn * 32 + 16 + r) * SH_LD + hi * 8]);
        fb1.u[1] = *(const uint4*)(&bB[(wn * 32 + 16 + r) * SH_LD + 16 + hi * 8]);

        acc00 = __builtin_amdgcn_wmma_f32_16x16x32_bf16(
            false, fa0.v, false, fb0.v, (short)0, acc00, false, false);
        acc01 = __builtin_amdgcn_wmma_f32_16x16x32_bf16(
            false, fa0.v, false, fb1.v, (short)0, acc01, false, false);
        acc10 = __builtin_amdgcn_wmma_f32_16x16x32_bf16(
            false, fa1.v, false, fb0.v, (short)0, acc10, false, false);
        acc11 = __builtin_amdgcn_wmma_f32_16x16x32_bf16(
            false, fa1.v, false, fb1.v, (short)0, acc11, false, false);
        __syncthreads();
#if defined(USE_ASYNC_LDS)
        cur ^= 1;
#endif
    }

    // ---- epilogue: bias (+GELU), bf16 or f32 store (M padded: no guards) ----
    const int row0 = mBlk + wm * 32 + hi * 8;     // C VGPR j -> row hi*8 + j
    const int row1 = row0 + 16;
    const int col0 = nBlk + wn * 32 + r;
    const int col1 = col0 + 16;
    const float bz0 = bias[col0];
    const float bz1 = bias[col1];
#pragma unroll
    for (int j = 0; j < 8; ++j) {
        float v00 = acc00[j] + bz0;
        float v01 = acc01[j] + bz1;
        float v10 = acc10[j] + bz0;
        float v11 = acc11[j] + bz1;
        if (ACT_GELU) {
            v00 = gelu_exact(v00); v01 = gelu_exact(v01);
            v10 = gelu_exact(v10); v11 = gelu_exact(v11);
        }
        if (OUT_BF16) {
            unsigned short* C = (unsigned short*)Cout;
            C[(size_t)(row0 + j) * ldc + col0] = f2bf(v00);
            C[(size_t)(row0 + j) * ldc + col1] = f2bf(v01);
            C[(size_t)(row1 + j) * ldc + col0] = f2bf(v10);
            C[(size_t)(row1 + j) * ldc + col1] = f2bf(v11);
        } else {
            float* C = (float*)Cout;
            C[(size_t)(row0 + j) * ldc + col0] = v00;
            C[(size_t)(row0 + j) * ldc + col1] = v01;
            C[(size_t)(row1 + j) * ldc + col0] = v10;
            C[(size_t)(row1 + j) * ldc + col1] = v11;
        }
    }
}

// ======================================================================
// small helper kernels
// ======================================================================
__global__ __launch_bounds__(256) void k_convert_bf16(
    const float* __restrict__ s, unsigned short* __restrict__ d, int n)
{
    int i = blockIdx.x * 256 + threadIdx.x;
    if (i < n) d[i] = f2bf(s[i]);
}

__global__ __launch_bounds__(256) void k_copy_f32(
    const float* __restrict__ s, float* __restrict__ d, int n)
{
    int i = blockIdx.x * 256 + threadIdx.x;
    if (i < n) d[i] = s[i];
}

// edge-enc layer1 folded into 50/2-row lookup tables:
//   A_tab = ne @ W0[:, 0:256]^T, R_tab = rel @ W0[:,256:512]^T, C_tab = ne @ W0[:,512:768]^T
__global__ __launch_bounds__(256) void k_tabs(
    const float* __restrict__ ne, const float* __restrict__ rel_emb,
    const float* __restrict__ W0,     // 256 x 768
    float* __restrict__ A_tab, float* __restrict__ C_tab, float* __restrict__ R_tab)
{
    int idx = blockIdx.x * 256 + threadIdx.x;   // 102 * 256 threads
    int r = idx >> 8, d = idx & 255;
    if (r < MAXNODE) {
        float a = 0.f;
        for (int k = 0; k < 256; ++k) a += ne[r * 256 + k] * W0[d * 768 + k];
        A_tab[r * 256 + d] = a;
    } else if (r < 2 * MAXNODE) {
        int n = r - MAXNODE; float a = 0.f;
        for (int k = 0; k < 256; ++k) a += ne[n * 256 + k] * W0[d * 768 + 512 + k];
        C_tab[n * 256 + d] = a;
    } else if (r < 2 * MAXNODE + 2) {
        int n = r - 2 * MAXNODE; float a = 0.f;
        for (int k = 0; k < 256; ++k) a += rel_emb[n * 256 + k] * W0[d * 768 + 256 + k];
        R_tab[n * 256 + d] = a;
    }
}

// T1[t,d] = GELU(A_tab[src] + R_tab[rel] + C_tab[tgt] + b0)  (edge tokens; 0 elsewhere)
__global__ __launch_bounds__(256) void k_edge_t1(
    const int* __restrict__ src, const int* __restrict__ rel, const int* __restrict__ tgt,
    const float* __restrict__ A_tab, const float* __restrict__ C_tab,
    const float* __restrict__ R_tab, const float* __restrict__ b0,
    unsigned short* __restrict__ T1)
{
    int idx = blockIdx.x * 256 + threadIdx.x;    // TPAD*256
    int t = idx >> 8, d = idx & 255;
    int b = t / (SEQL + 1), j = t - b * (SEQL + 1);
    float v = 0.f;
    if (t < T_REAL && j < SEQL) {
        int e = b * SEQL + j;
        int s  = min(max(src[e], 0), MAXNODE - 1);
        int rl = rel[e];
        int tg = min(max(tgt[e], 0), MAXNODE - 1);
        v = gelu_exact(A_tab[s * 256 + d] + R_tab[rl * 256 + d] + C_tab[tg * 256 + d] + b0[d]);
    }
    T1[idx] = f2bf(v);
}

// full query-token MLP (32 tokens) in f32, overwrites X rows b*2049+2048
__global__ __launch_bounds__(256) void k_query(
    const float* __restrict__ ne,
    const float* __restrict__ w0, const float* __restrict__ b0,   // 256x512, 256
    const float* __restrict__ w1, const float* __restrict__ b1,   // 256x256, 256
    const int* __restrict__ qsrc, const int* __restrict__ qtgt,
    unsigned short* __restrict__ XS)
{
    __shared__ float qin[512];
    __shared__ float h[256];
    int b = blockIdx.x, tid = threadIdx.x;
    int qs = min(max(qsrc[b], 0), MAXNODE - 1);
    int qt = min(max(qtgt[b], 0), MAXNODE - 1);
    qin[tid]       = ne[qs * 256 + tid];
    qin[256 + tid] = ne[qt * 256 + tid];
    __syncthreads();
    float a = b0[tid];
    for (int k = 0; k < 512; ++k) a += qin[k] * w0[tid * 512 + k];
    h[tid] = gelu_exact(a);
    __syncthreads();
    float a2 = b1[tid];
    for (int k = 0; k < 256; ++k) a2 += h[k] * w1[tid * 256 + k];
    XS[((size_t)(b * (SEQL + 1) + SEQL)) * 512 + tid] = f2bf(a2);
}

// routing: argmax(X@group^T), argmax(X@slot^T) -> flat idx; plus gathered src ids
__global__ __launch_bounds__(256) void k_route(
    const unsigned short* __restrict__ XS,
    const float* __restrict__ gproj, const float* __restrict__ sproj,
    const int* __restrict__ src, const int* __restrict__ qsrc,
    int* __restrict__ flat, int* __restrict__ srcid)
{
    int lane = threadIdx.x & 31, wave = threadIdx.x >> 5;
    int t = blockIdx.x * 8 + wave;
    if (t >= T_REAL) return;
    const unsigned short* xr = XS + (size_t)t * 512;
    const float* pr = (lane < 16) ? (gproj + lane * 256) : (sproj + (lane - 16) * 256);
    float s = 0.f;
    for (int d = 0; d < 256; ++d) s += bf2f(xr[d]) * pr[d];
    int idx = lane & 15;
    // argmax within each 16-lane half (first-max tie break)
    for (int m = 1; m < 16; m <<= 1) {
        float os = __shfl_xor(s, m);
        int   oi = __shfl_xor(idx, m);
        if (os > s || (os == s && oi < idx)) { s = os; idx = oi; }
    }
    int g  = __shfl(idx, 0);
    int sl = __shfl(idx, 16);
    if (lane == 0) {
        flat[t] = g * KSLOT + sl;
        int b = t / (SEQL + 1), j = t - b * (SEQL + 1);
        int sv = (j < SEQL) ? src[b * SEQL + j] : qsrc[b];
        srcid[t] = min(max(sv, 0), NSLOT - 1);
    }
}

__global__ __launch_bounds__(256) void k_S_init(
    const float* __restrict__ S_init, float* __restrict__ S)
{
    int idx = blockIdx.x * 256 + threadIdx.x;   // 32*65536
    S[idx] = S_init[idx & 65535];
}

// XS[:,256:512] = bf16(S[b, srcid[t]])
__global__ __launch_bounds__(256) void k_gather_src(
    const float* __restrict__ S, const int* __restrict__ srcid,
    unsigned short* __restrict__ XS)
{
    int idx = blockIdx.x * 256 + threadIdx.x;   // TPAD*256
    int t = idx >> 8, d = idx & 255;
    float v = 0.f;
    if (t < T_REAL) {
        int b = t / (SEQL + 1);
        v = S[((size_t)(b * NSLOT + srcid[t])) * 256 + d];
    }
    XS[(size_t)t * 512 + 256 + d] = f2bf(v);
}

// gate = sigmoid(H1[t,512:768] . wg1 + gb); scatter-add gate*Mout into incoming
__global__ __launch_bounds__(256) void k_gate_scatter(
    const unsigned short* __restrict__ H1, const float* __restrict__ Mout,
    const float* __restrict__ wg1, const float* __restrict__ gb1,
    const int* __restrict__ flat, float* __restrict__ incoming)
{
    int lane = threadIdx.x & 31, wave = threadIdx.x >> 5;
    int t = blockIdx.x * 8 + wave;
    if (t >= T_REAL) return;
    const unsigned short* hr = H1 + (size_t)t * 768 + 512;
    float p = 0.f;
#pragma unroll
    for (int i = 0; i < 8; ++i) {
        int d = lane + 32 * i;
        p += bf2f(hr[d]) * wg1[d];
    }
    for (int m = 16; m > 0; m >>= 1) p += __shfl_xor(p, m);
    float gate = 1.0f / (1.0f + expf(-(p + gb1[0])));
    int b = t / (SEQL + 1);
    float* dst = incoming + ((size_t)(b * NSLOT + flat[t])) * 256;
    const float* mr = Mout + (size_t)t * 256;
#pragma unroll
    for (int i = 0; i < 8; ++i) {
        int d = lane + 32 * i;
        atomicAdd(&dst[d], mr[d] * gate);
    }
}

// SI = bf16(cat[S, incoming])  (8192 x 512)
__global__ __launch_bounds__(256) void k_cat_SI(
    const float* __restrict__ S, const float* __restrict__ incoming,
    unsigned short* __restrict__ SI)
{
    int idx = blockIdx.x * 256 + threadIdx.x;   // 8192*512
    int r = idx >> 9, c = idx & 511;
    float v = (c < 256) ? S[(size_t)r * 256 + c] : incoming[(size_t)r * 256 + (c - 256)];
    SI[idx] = f2bf(v);
}

// S = LN(S + dS) * g + b  (wave per slot row; population variance)
__global__ __launch_bounds__(256) void k_ln(
    float* __restrict__ S, const float* __restrict__ dS,
    const float* __restrict__ g, const float* __restrict__ bb)
{
    int lane = threadIdx.x & 31, wave = threadIdx.x >> 5;
    int row = blockIdx.x * 8 + wave;
    float* sr = S + (size_t)row * 256;
    const float* dr = dS + (size_t)row * 256;
    float v[8], sum = 0.f, sq = 0.f;
#pragma unroll
    for (int i = 0; i < 8; ++i) {
        int d = lane + 32 * i;
        v[i] = sr[d] + dr[d];
        sum += v[i]; sq += v[i] * v[i];
    }
    for (int m = 16; m > 0; m >>= 1) { sum += __shfl_xor(sum, m); sq += __shfl_xor(sq, m); }
    float mu = sum * (1.0f / 256.0f);
    float var = sq * (1.0f / 256.0f) - mu * mu;
    float rs = rsqrtf(var + 1e-5f);
#pragma unroll
    for (int i = 0; i < 8; ++i) {
        int d = lane + 32 * i;
        sr[d] = (v[i] - mu) * rs * g[d] + bb[d];
    }
}

// head MLP: out[b] = (cat[S[b,qs],S[b,qt]] -> 512->256 GELU -> 2)
__global__ __launch_bounds__(256) void k_head(
    const float* __restrict__ S,
    const float* __restrict__ w0, const float* __restrict__ b0,
    const float* __restrict__ w1, const float* __restrict__ b1,
    const int* __restrict__ qsrc, const int* __restrict__ qtgt,
    float* __restrict__ out)
{
    __shared__ float hin[512];
    __shared__ float h[256];
    int b = blockIdx.x, tid = threadIdx.x;
    int qs = min(max(qsrc[b], 0), NSLOT - 1);
    int qt = min(max(qtgt[b], 0), NSLOT - 1);
    hin[tid]       = S[((size_t)(b * NSLOT + qs)) * 256 + tid];
    hin[256 + tid] = S[((size_t)(b * NSLOT + qt)) * 256 + tid];
    __syncthreads();
    float a = b0[tid];
    for (int k = 0; k < 512; ++k) a += hin[k] * w0[tid * 512 + k];
    h[tid] = gelu_exact(a);
    __syncthreads();
    if (tid < 2) {
        float a2 = b1[tid];
        for (int k = 0; k < 256; ++k) a2 += h[k] * w1[tid * 256 + k];
        out[b * 2 + tid] = a2;
    }
}

// ======================================================================
// host launcher
// ======================================================================
enum {
    IN_NODE_EMBED = 0, IN_REL_EMBED,
    IN_EE0_W, IN_EE0_B, IN_EE1_W, IN_EE1_B,
    IN_QE0_W, IN_QE0_B, IN_QE1_W, IN_QE1_B,
    IN_GROUP_PROJ, IN_SLOT_PROJ,
    IN_MSG0_W, IN_MSG0_B, IN_MSG1_W, IN_MSG1_B,
    IN_GATE0_W, IN_GATE0_B, IN_GATE1_W, IN_GATE1_B,
    IN_UPD0_W, IN_UPD0_B, IN_UPD1_W, IN_UPD1_B,
    IN_LN_G, IN_LN_B, IN_S_INIT,
    IN_HEAD0_W, IN_HEAD0_B, IN_HEAD1_W, IN_HEAD1_B,
    IN_SRC, IN_REL, IN_TGT, IN_MASK, IN_QSRC, IN_QTGT
};

extern "C" void kernel_launch(void* const* d_in, const int* in_sizes, int n_in,
                              void* d_out, int out_size, void* d_ws, size_t ws_size,
                              hipStream_t stream)
{
    (void)in_sizes; (void)n_in; (void)out_size; (void)ws_size;

    const float* ne    = (const float*)d_in[IN_NODE_EMBED];
    const float* relE  = (const float*)d_in[IN_REL_EMBED];
    const float* ee0w  = (const float*)d_in[IN_EE0_W];
    const float* ee0b  = (const float*)d_in[IN_EE0_B];
    const float* ee1w  = (const float*)d_in[IN_EE1_W];
    const float* ee1b  = (const float*)d_in[IN_EE1_B];
    const float* qe0w  = (const float*)d_in[IN_QE0_W];
    const float* qe0b  = (const float*)d_in[IN_QE0_B];
    const float* qe1w  = (const float*)d_in[IN_QE1_W];
    const float* qe1b  = (const float*)d_in[IN_QE1_B];
    const float* gproj = (const float*)d_in[IN_GROUP_PROJ];
    const float* sproj = (const float*)d_in[IN_SLOT_PROJ];
    const float* m0w   = (const float*)d_in[IN_MSG0_W];
    const float* m0b   = (const float*)d_in[IN_MSG0_B];
    const float* m1w   = (const float*)d_in[IN_MSG1_W];
    const float* m1b   = (const float*)d_in[IN_MSG1_B];
    const float* g0w   = (const float*)d_in[IN_GATE0_W];
    const float* g0b   = (const float*)d_in[IN_GATE0_B];
    const float* g1w   = (const float*)d_in[IN_GATE1_W];
    const float* g1b   = (const float*)d_in[IN_GATE1_B];
    const float* u0w   = (const float*)d_in[IN_UPD0_W];
    const float* u0b   = (const float*)d_in[IN_UPD0_B];
    const float* u1w   = (const float*)d_in[IN_UPD1_W];
    const float* u1b   = (const float*)d_in[IN_UPD1_B];
    const float* lng   = (const float*)d_in[IN_LN_G];
    const float* lnb   = (const float*)d_in[IN_LN_B];
    const float* Sini  = (const float*)d_in[IN_S_INIT];
    const float* h0w   = (const float*)d_in[IN_HEAD0_W];
    const float* h0b   = (const float*)d_in[IN_HEAD0_B];
    const float* h1w   = (const float*)d_in[IN_HEAD1_W];
    const float* h1b   = (const float*)d_in[IN_HEAD1_B];
    const int*   src   = (const int*)d_in[IN_SRC];
    const int*   rel   = (const int*)d_in[IN_REL];
    const int*   tgt   = (const int*)d_in[IN_TGT];
    const int*   qsrc  = (const int*)d_in[IN_QSRC];
    const int*   qtgt  = (const int*)d_in[IN_QTGT];
    float* out = (float*)d_out;

    // ---------- workspace carve-up ----------
    char* ws = (char*)d_ws;
    size_t off = 0;
    auto carve = [&](size_t bytes) -> void* {
        void* p = ws + off;
        off += (bytes + 255) & ~(size_t)255;
        return p;
    };
    unsigned short* XS  = (unsigned short*)carve((size_t)TPAD * 512 * 2); // [X | src_state]
    unsigned short* H1  = (unsigned short*)carve((size_t)TPAD * 768 * 2); // msg+gate hidden (T1 aliases here)
    float*          Mout= (float*)carve((size_t)TPAD * 256 * 4);
    float*          S   = (float*)carve((size_t)SROWS * 256 * 4);
    float*          INC = (float*)carve((size_t)SROWS * 256 * 4);
    unsigned short* SI  = (unsigned short*)carve((size_t)SROWS * 512 * 2);
    unsigned short* U1  = (unsigned short*)carve((size_t)SROWS * 512 * 2);
    float*          DSb = (float*)carve((size_t)SROWS * 256 * 4);
    unsigned short* W0cat = (unsigned short*)carve((size_t)768 * 512 * 2); // [msg0.w ; gate0.w]
    unsigned short* WM1 = (unsigned short*)carve((size_t)256 * 512 * 2);
    unsigned short* WU0 = (unsigned short*)carve((size_t)512 * 512 * 2);
    unsigned short* WU1 = (unsigned short*)carve((size_t)256 * 512 * 2);
    unsigned short* WE1 = (unsigned short*)carve((size_t)256 * 256 * 2);
    float*          B0cat = (float*)carve(768 * 4);
    float*          ATAB = (float*)carve((size_t)MAXNODE * 256 * 4);
    float*          CTAB = (float*)carve((size_t)MAXNODE * 256 * 4);
    float*          RTAB = (float*)carve((size_t)2 * 256 * 4);
    int*            FLAT = (int*)carve((size_t)TPAD * 4);
    int*            SRCID= (int*)carve((size_t)TPAD * 4);
    unsigned short* T1 = H1;   // T1 (TPAD x 256 bf16) consumed before H1 is first written

    // ---------- one-time prep ----------
    k_convert_bf16<<<(512 * 512 + 255) / 256, 256, 0, stream>>>(m0w, W0cat, 512 * 512);
    k_convert_bf16<<<(256 * 512 + 255) / 256, 256, 0, stream>>>(g0w, W0cat + 512 * 512, 256 * 512);
    k_convert_bf16<<<(256 * 512 + 255) / 256, 256, 0, stream>>>(m1w, WM1, 256 * 512);
    k_convert_bf16<<<(512 * 512 + 255) / 256, 256, 0, stream>>>(u0w, WU0, 512 * 512);
    k_convert_bf16<<<(256 * 512 + 255) / 256, 256, 0, stream>>>(u1w, WU1, 256 * 512);
    k_convert_bf16<<<(256 * 256 + 255) / 256, 256, 0, stream>>>(ee1w, WE1, 256 * 256);
    k_copy_f32<<<2, 256, 0, stream>>>(m0b, B0cat, 512);
    k_copy_f32<<<1, 256, 0, stream>>>(g0b, B0cat + 512, 256);

    // edge layer1 as lookup tables (vocab 50 / 2): kills a 26-GFLOP GEMM
    k_tabs<<<2 * MAXNODE + 2, 256, 0, stream>>>(ne, relE, ee0w, ATAB, CTAB, RTAB);
    k_edge_t1<<<TPAD, 256, 0, stream>>>(src, rel, tgt, ATAB, CTAB, RTAB, ee0b, T1);

    // X = T1 @ We1^T + b  -> XS[:, 0:256] (bf16, ldc=512)
    gemm_bf16_wmma<false, true><<<dim3(TPAD / 128, 256 / 64), 256, 0, stream>>>(
        T1, 256, WE1, 256, ee1b, XS, 512, TPAD, 256, 256);
    // query tokens: full f32 MLP overwrite of 32 rows
    k_query<<<BATCH, 256, 0, stream>>>(ne, qe0w, qe0b, qe1w, qe1b, qsrc, qtgt, XS);
    // routing (static across steps) + gathered src ids
    k_route<<<(T_REAL + 7) / 8, 256, 0, stream>>>(XS, gproj, sproj, src, qsrc, FLAT, SRCID);
    // S <- broadcast S_init
    k_S_init<<<(BATCH * NSLOT * 256) / 256, 256, 0, stream>>>(Sini, S);

    // ---------- 24 recurrence steps ----------
    for (int step = 0; step < NSTEPS; ++step) {
        // XS right half = bf16(S[b, src_id])
        k_gather_src<<<TPAD, 256, 0, stream>>>(S, SRCID, XS);
        // H1 = GELU(XS @ [msg0;gate0]^T + b)   (M=TPAD, N=768, K=512)
        gemm_bf16_wmma<true, true><<<dim3(TPAD / 128, 768 / 64), 256, 0, stream>>>(
            XS, 512, W0cat, 512, B0cat, H1, 768, TPAD, 768, 512);
        // Mout = H1[:, :512] @ msg1^T + b      (M=TPAD, N=256, K=512)
        gemm_bf16_wmma<false, false><<<dim3(TPAD / 128, 256 / 64), 256, 0, stream>>>(
            H1, 768, WM1, 512, m1b, Mout, 256, TPAD, 256, 512);
        // incoming = scatter-add(sigmoid(gate) * Mout)
        (void)hipMemsetAsync(INC, 0, (size_t)SROWS * 256 * 4, stream);
        k_gate_scatter<<<(T_REAL + 7) / 8, 256, 0, stream>>>(H1, Mout, g1w, g1b, FLAT, INC);
        // SI = bf16([S, incoming]); U1 = GELU(SI @ upd0^T + b); dS = U1 @ upd1^T + b
        k_cat_SI<<<(SROWS * 512) / 256, 256, 0, stream>>>(S, INC, SI);
        gemm_bf16_wmma<true, true><<<dim3(SROWS / 128, 512 / 64), 256, 0, stream>>>(
            SI, 512, WU0, 512, u0b, U1, 512, SROWS, 512, 512);
        gemm_bf16_wmma<false, false><<<dim3(SROWS / 128, 256 / 64), 256, 0, stream>>>(
            U1, 512, WU1, 512, u1b, DSb, 256, SROWS, 256, 512);
        // S = LN(S + dS)
        k_ln<<<SROWS / 8, 256, 0, stream>>>(S, DSb, lng, lnb);
    }

    // ---------- head ----------
    k_head<<<BATCH, 256, 0, stream>>>(S, h0w, h0b, h1w, h1b, qsrc, qtgt, out);
}
